// EdgeProcessor_22840636080714
// MI455X (gfx1250) — matrix operand order repaired and checked by reading, not compiled, not run on hardware
//
#include <hip/hip_runtime.h>
#include <stdint.h>

typedef __attribute__((ext_vector_type(16))) __bf16 v16bf;
typedef __attribute__((ext_vector_type(8)))  float  v8f;

#define WAVES 16
#define BLOCK (WAVES * 32)

// ---- LDS layout (bytes) ----
#define W_ROW_B      272                   // 128 bf16 + 8 pad -> 4-bank step, conflict-free b128
#define W0_OFF       0
#define W0_B         (384 * W_ROW_B)       // 104448
#define W1_OFF       (W0_OFF + W0_B)
#define W12_B        (128 * W_ROW_B)       // 34816
#define W2_OFF       (W1_OFF + W12_B)
#define STAGE_OFF    (W2_OFF + W12_B)      // 174080
#define STG_ROW_B    272                   // h staging row: 128 bf16 + 8 pad
#define STG_B        (16 * STG_ROW_B)      // 4352 per wave
#define SMEM_TOTAL   (STAGE_OFF + WAVES * STG_B)   // 243712 B (< 320 KB WGP LDS)

struct alignas(16) U128 { uint32_t a, b, c, d; };
struct alignas(16) F4   { float x, y, z, w; };
struct alignas(32) Frag { U128 lo, hi; };   // 32 B == v16bf

// one v_perm_b32: (bits(b) & 0xFFFF0000) | (bits(a) >> 16)  -> packed {bf16(b), bf16(a)}
__device__ __forceinline__ uint32_t pk_bf16_fast(float a, float b) {
    return __builtin_amdgcn_perm(__float_as_uint(b), __float_as_uint(a), 0x07060302u);
}
// round-to-nearest-even pack (cold path: weight conversion only)
__device__ __forceinline__ uint32_t pk_bf16_rne(float a, float b) {
    uint32_t ua = __float_as_uint(a), ub = __float_as_uint(b);
    ua = (ua + 0x7FFFu + ((ua >> 16) & 1u)) >> 16;
    ub = (ub + 0x7FFFu + ((ub >> 16) & 1u)) >> 16;
    return (ub << 16) | (ua & 0xFFFFu);
}

__global__ __launch_bounds__(BLOCK)
__attribute__((amdgpu_waves_per_eu(4)))    // <=256 VGPRs; rolled K-loops keep us far below it
void edge_mlp_wmma(const float* __restrict__ node, const long long* __restrict__ em,
                   const float* __restrict__ ef,
                   const float* __restrict__ W0, const float* __restrict__ b0,
                   const float* __restrict__ W1, const float* __restrict__ b1,
                   const float* __restrict__ W2, const float* __restrict__ b2,
                   const float* __restrict__ gamma, const float* __restrict__ beta,
                   float* __restrict__ out, int Ne)
{
    extern __shared__ char smem[];
    const int tid  = threadIdx.x;
    const int lane = tid & 31;
    const int m    = lane & 15;     // A row / C col index
    const int g    = lane >> 4;     // lane half

    // ---- convert weights f32 -> bf16 (RNE) into padded LDS, once per workgroup ----
    for (int i = tid; i < 384 * 64; i += BLOCK) {
        int k = i >> 6, p = i & 63;
        const float* s = W0 + k * 128 + p * 2;
        *(uint32_t*)(smem + W0_OFF + k * W_ROW_B + p * 4) = pk_bf16_rne(s[0], s[1]);
    }
    for (int i = tid; i < 128 * 64; i += BLOCK) {
        int k = i >> 6, p = i & 63;
        const float* s1 = W1 + k * 128 + p * 2;
        const float* s2 = W2 + k * 128 + p * 2;
        *(uint32_t*)(smem + W1_OFF + k * W_ROW_B + p * 4) = pk_bf16_rne(s1[0], s1[1]);
        *(uint32_t*)(smem + W2_OFF + k * W_ROW_B + p * 4) = pk_bf16_rne(s2[0], s2[1]);
    }
    __syncthreads();

    // per-lane epilogue params (col = t*16 + m)
    float pb0[8], pb1[8], pb2[8], pgm[8], pbt[8];
#pragma unroll
    for (int t = 0; t < 8; ++t) {
        int n = t * 16 + m;
        pb0[t] = b0[n]; pb1[t] = b1[n]; pb2[t] = b2[n];
        pgm[t] = gamma[n]; pbt[t] = beta[n];
    }

    char* stage = smem + STAGE_OFF + (tid >> 5) * STG_B;
    const int nTiles = (Ne + 15) >> 4;
    const int wave   = blockIdx.x * WAVES + (tid >> 5);
    const int nWave  = gridDim.x * WAVES;
    const v8f vzero = {0.f, 0.f, 0.f, 0.f, 0.f, 0.f, 0.f, 0.f};

    for (int tile = wave; tile < nTiles; tile += nWave) {
        const int ebase = tile * 16;

        // per-lane source row pointers for edge m of this tile
        int eidx = ebase + m; if (eidx >= Ne) eidx = Ne - 1;
        const float* rp[3];
        rp[0] = node + (size_t)em[eidx] * 128;        // sender row
        rp[1] = node + (size_t)em[Ne + eidx] * 128;   // receiver row
        rp[2] = ef   + (size_t)eidx * 128;            // edge row

        v8f acc[8];
#pragma unroll
        for (int t = 0; t < 8; ++t) acc[t] = vzero;

        // ---- layer 0: [16x384] @ [384x128]; A built straight from global ----
#pragma unroll
        for (int part = 0; part < 3; ++part) {
            const float* rpp = rp[part];
#pragma unroll 1                       // rolled: bounds live B-fragments, no spills
            for (int q = 0; q < 4; ++q) {
                const int off = q * 32 + g * 8;
                F4 f0 = *(const F4*)(rpp + off);
                F4 f1 = *(const F4*)(rpp + off + 4);
                F4 f2 = *(const F4*)(rpp + off + 16);
                F4 f3 = *(const F4*)(rpp + off + 20);
                Frag fa;
                fa.lo = U128{pk_bf16_fast(f0.x, f0.y), pk_bf16_fast(f0.z, f0.w),
                             pk_bf16_fast(f1.x, f1.y), pk_bf16_fast(f1.z, f1.w)};
                fa.hi = U128{pk_bf16_fast(f2.x, f2.y), pk_bf16_fast(f2.z, f2.w),
                             pk_bf16_fast(f3.x, f3.y), pk_bf16_fast(f3.z, f3.w)};
                v16bf av = __builtin_bit_cast(v16bf, fa);
                const char* brow = smem + W0_OFF + ((part * 4 + q) * 32 + g * 8) * W_ROW_B + m * 2;
#pragma unroll
                for (int t = 0; t < 8; ++t) {
                    v16bf bv;
#pragma unroll
                    for (int e = 0; e < 16; ++e)
                        bv[e] = *(const __bf16*)(brow + ((e & 7) + ((e >> 3) << 4)) * W_ROW_B + t * 32);
                    acc[t] = __builtin_amdgcn_wmma_f32_16x16x32_bf16(
                        false, av, false, bv, (short)0, acc[t], false, false);
                }
            }
        }
        // bias + relu -> stage h as bf16 (DS in-order within wave: no barrier)
#pragma unroll
        for (int t = 0; t < 8; ++t)
#pragma unroll
            for (int r = 0; r < 8; ++r) {
                float f = fmaxf(acc[t][r] + pb0[t], 0.f);
                *(uint16_t*)(stage + (r + 8 * g) * STG_ROW_B + (t * 16 + m) * 2) =
                    (uint16_t)(__float_as_uint(f) >> 16);   // b16_d16_hi store
            }

        // ---- layer 1: [16x128] @ [128x128] ----
#pragma unroll
        for (int t = 0; t < 8; ++t) acc[t] = vzero;
#pragma unroll 1
        for (int s = 0; s < 4; ++s) {
            Frag fa;
            const char* ab = stage + m * STG_ROW_B + (s * 32 + g * 8) * 2;
            fa.lo = *(const U128*)ab; fa.hi = *(const U128*)(ab + 32);
            v16bf av = __builtin_bit_cast(v16bf, fa);
            const char* brow = smem + W1_OFF + (s * 32 + g * 8) * W_ROW_B + m * 2;
#pragma unroll
            for (int t = 0; t < 8; ++t) {
                v16bf bv;
#pragma unroll
                for (int e = 0; e < 16; ++e)
                    bv[e] = *(const __bf16*)(brow + ((e & 7) + ((e >> 3) << 4)) * W_ROW_B + t * 32);
                acc[t] = __builtin_amdgcn_wmma_f32_16x16x32_bf16(
                    false, av, false, bv, (short)0, acc[t], false, false);
            }
        }
#pragma unroll
        for (int t = 0; t < 8; ++t)
#pragma unroll
            for (int r = 0; r < 8; ++r) {
                float f = fmaxf(acc[t][r] + pb1[t], 0.f);
                *(uint16_t*)(stage + (r + 8 * g) * STG_ROW_B + (t * 16 + m) * 2) =
                    (uint16_t)(__float_as_uint(f) >> 16);
            }

        // ---- layer 2: [16x128] @ [128x128] ----
#pragma unroll
        for (int t = 0; t < 8; ++t) acc[t] = vzero;
#pragma unroll 1
        for (int s = 0; s < 4; ++s) {
            Frag fa;
            const char* ab = stage + m * STG_ROW_B + (s * 32 + g * 8) * 2;
            fa.lo = *(const U128*)ab; fa.hi = *(const U128*)(ab + 32);
            v16bf av = __builtin_bit_cast(v16bf, fa);
            const char* brow = smem + W2_OFF + (s * 32 + g * 8) * W_ROW_B + m * 2;
#pragma unroll
            for (int t = 0; t < 8; ++t) {
                v16bf bv;
#pragma unroll
                for (int e = 0; e < 16; ++e)
                    bv[e] = *(const __bf16*)(brow + ((e & 7) + ((e >> 3) << 4)) * W_ROW_B + t * 32);
                acc[t] = __builtin_amdgcn_wmma_f32_16x16x32_bf16(
                    false, av, false, bv, (short)0, acc[t], false, false);
            }
        }

        // ---- bias + LayerNorm + store f32 ----
        // C layout: lane half g owns rows r+8g (r=0..7), col = t*16 + m.
#pragma unroll
        for (int r = 0; r < 8; ++r) {
            float s1 = 0.f, s2 = 0.f;
#pragma unroll
            for (int t = 0; t < 8; ++t) {
                float f = acc[t][r] + pb2[t];
                s1 += f; s2 += f * f;
            }
#pragma unroll
            for (int off = 1; off < 16; off <<= 1) {   // xor masks 1,2,4,8 stay in-half
                s1 += __shfl_xor(s1, off, 32);
                s2 += __shfl_xor(s2, off, 32);
            }
            float mean = s1 * (1.f / 128.f);
            float var  = s2 * (1.f / 128.f) - mean * mean;
            float inv  = rsqrtf(var + 1e-5f);
            int row = ebase + r + 8 * g;
            if (row < Ne) {
                float* orow = out + (size_t)row * 128;
#pragma unroll
                for (int t = 0; t < 8; ++t) {
                    float f = acc[t][r] + pb2[t];
                    orow[t * 16 + m] = (f - mean) * inv * pgm[t] + pbt[t];
                }
            }
        }
    }
}

extern "C" void kernel_launch(void* const* d_in, const int* in_sizes, int n_in,
                              void* d_out, int out_size, void* d_ws, size_t ws_size,
                              hipStream_t stream) {
    const float*     node  = (const float*)d_in[0];
    const long long* em    = (const long long*)d_in[1];  // int64 [2,E]
    const float*     ef    = (const float*)d_in[2];
    const float*     W0    = (const float*)d_in[3];
    const float*     b0    = (const float*)d_in[4];
    const float*     W1    = (const float*)d_in[5];
    const float*     b1    = (const float*)d_in[6];
    const float*     W2    = (const float*)d_in[7];
    const float*     b2    = (const float*)d_in[8];
    const float*     gamma = (const float*)d_in[9];
    const float*     beta  = (const float*)d_in[10];
    float* out = (float*)d_out;
    const int Ne = in_sizes[1] / 2;

    (void)hipFuncSetAttribute((const void*)edge_mlp_wmma,
                              hipFuncAttributeMaxDynamicSharedMemorySize, SMEM_TOTAL);
    const int blocks = 512;   // grid-stride over E/16 = 31250 wave-tiles
    edge_mlp_wmma<<<blocks, BLOCK, SMEM_TOTAL, stream>>>(
        node, em, ef, W0, b0, W1, b1, W2, b2, gamma, beta, out, Ne);
}
